// MoE_36661840839151
// MI455X (gfx1250) — matrix operand order repaired and checked by reading, not compiled, not run on hardware
//
#include <hip/hip_runtime.h>
#include <math.h>

#define HIDDEN 768
#define INTER  2048
#define NEXP   8
#define NTOK   8192
#define LIMIT  7.0f

#define MTILE  32
#define NWAVES 8
#define XSTR   (HIDDEN + 8)        // bf16 elems, row-major staged x tile
#define HTSTRIDE (512 + 16)        // elems per swizzled 16x32 h tile (16-elem pad)

typedef __attribute__((ext_vector_type(16))) __bf16        v16bf;
typedef __attribute__((ext_vector_type(8)))  float         v8f;
typedef __attribute__((ext_vector_type(8)))  unsigned int  v8u;

// ---------- fp32 -> bf16 (round-nearest-even); only used in pre-pass / epilogues ----------
__device__ inline unsigned short f2bf(float f) {
  unsigned int u = __builtin_bit_cast(unsigned int, f);
  u += 0x7fffu + ((u >> 16) & 1u);
  return (unsigned short)(u >> 16);
}
__device__ inline unsigned int pack2(float a, float b) {
  return (unsigned int)f2bf(a) | ((unsigned int)f2bf(b) << 16);
}

// =======================================================================
// Fragment layouts (ISA 7.12.2, wave32):
//  A 16x32 bf16: lane l -> row m = l&15; K set {kh..kh+7, 16+kh..16+kh+7}, kh = 8*(l>>4).
//    Within the lane's 16 values, j<8 -> K = kh+j ; j>=8 -> K = 16+kh+(j-8).
//  B 32x16 bf16: lane l -> col n = l&15; same K set/order as A (kh = 8*(l>>4)).
// Swizzled storage: tile of 512 bf16, element index = lane*16 + j  (one 32B vector per lane).

__device__ inline v16bf frag_from(const unsigned short* p, int lane) {
  v8u u = *(const v8u*)(p + lane * 16);          // 32B contiguous per lane
  return __builtin_bit_cast(v16bf, u);
}

// A-fragment from row-major bf16 LDS tile (used for x; 8 dword reads)
__device__ inline v16bf load_a_rowmajor(const unsigned short* s, int stride,
                                        int m0, int k0, int lane) {
  const unsigned short* base = s + (m0 + (lane & 15)) * stride;
  const int kh = (lane >> 4) << 3;
  v8u u;
#pragma unroll
  for (int v = 0; v < 4; ++v)
    u[v] = *(const unsigned int*)(base + k0 + kh + 2 * v);
#pragma unroll
  for (int v = 0; v < 4; ++v)
    u[4 + v] = *(const unsigned int*)(base + k0 + 16 + kh + 2 * v);
  return __builtin_bit_cast(v16bf, u);
}

__device__ inline v8f wmma_bf16(v16bf a, v16bf b, v8f c) {
  return __builtin_amdgcn_wmma_f32_16x16x32_bf16(false, a, false, b,
                                                 (short)0, c, false, false);
}

// LDS byte offset of a generic pointer to __shared__ (low 32 bits of the
// shared aperture address are the LDS offset on amdgcn).
__device__ inline unsigned lds_off(const void* p) {
  return (unsigned)(unsigned long long)p;
}

// =======================================================================
__global__ __launch_bounds__(256) void zero_out_kernel(float* __restrict__ out, int n) {
  int i = blockIdx.x * 256 + threadIdx.x;
  if (i < n) out[i] = 0.0f;
}

// x fp32 -> bf16, row-major (float4 -> 2 dwords)
__global__ __launch_bounds__(256) void cvt_x_kernel(const float* __restrict__ x,
                                                    unsigned short* __restrict__ x16) {
  int i = blockIdx.x * 256 + threadIdx.x;        // one float4 per thread
  if (i >= NTOK * HIDDEN / 4) return;
  float4 v = *(const float4*)(x + (size_t)i * 4);
  unsigned short* d = x16 + (size_t)i * 4;
  *(unsigned int*)(d)     = pack2(v.x, v.y);
  *(unsigned int*)(d + 2) = pack2(v.z, v.w);
}

// Weight fp32 [N][K] -> bf16 swizzled B-fragment tiles:
//   tile = nt*(K/32)+kt ; dst[tile*512 + lane*16 + j] = W[nt*16+(lane&15)][kt*32+8*(lane>>4)+j+(j>=8?8:0)]
__global__ __launch_bounds__(256) void swz_w_kernel(const float* __restrict__ W,
                                                    unsigned short* __restrict__ dst,
                                                    int N, int K) {
  int job = blockIdx.x * 256 + threadIdx.x;      // one (tile, lane)
  int total = (N >> 4) * (K >> 5) * 32;
  if (job >= total) return;
  int lane = job & 31;
  int tile = job >> 5;
  int kt = tile % (K >> 5);
  int nt = tile / (K >> 5);
  const float* src = W + (size_t)(nt * 16 + (lane & 15)) * K + kt * 32 + ((lane >> 4) << 3);
  unsigned short* d = dst + (size_t)tile * 512 + lane * 16;
#pragma unroll
  for (int j = 0; j < 16; j += 2) {
    float2 f = *(const float2*)(src + j + (j >= 8 ? 8 : 0));
    *(unsigned int*)(d + j) = pack2(f.x, f.y);
  }
}

// one thread per token; router_w staged in LDS (24 KB)
__global__ __launch_bounds__(256) void router_kernel(
    const float* __restrict__ x, const float* __restrict__ rw,
    int* __restrict__ topk_idx, float* __restrict__ topk_w) {
  __shared__ float rws[NEXP * HIDDEN];
  const int tid = threadIdx.x;
  for (int i = tid; i < NEXP * HIDDEN; i += 256) rws[i] = rw[i];
  __syncthreads();

  const int t = blockIdx.x * 256 + tid;
  if (t >= NTOK) return;
  const float* xr = x + (size_t)t * HIDDEN;

  float logits[NEXP];
#pragma unroll
  for (int e = 0; e < NEXP; ++e) logits[e] = 0.0f;
  for (int k = 0; k < HIDDEN; k += 4) {
    float4 xv = *(const float4*)(xr + k);
#pragma unroll
    for (int e = 0; e < NEXP; ++e) {
      const float* r = rws + e * HIDDEN + k;
      logits[e] += xv.x * r[0] + xv.y * r[1] + xv.z * r[2] + xv.w * r[3];
    }
  }
  int i0 = 0; float v0 = logits[0];
#pragma unroll
  for (int e = 1; e < NEXP; ++e)
    if (logits[e] > v0) { v0 = logits[e]; i0 = e; }
  int i1 = -1; float v1 = -INFINITY;
#pragma unroll
  for (int e = 0; e < NEXP; ++e)
    if (e != i0 && logits[e] > v1) { v1 = logits[e]; i1 = e; }

  float w0 = __builtin_amdgcn_rcpf(1.0f + __expf(v1 - v0));  // softmax over 2
  topk_idx[2 * t] = i0;  topk_idx[2 * t + 1] = i1;
  topk_w[2 * t] = w0;    topk_w[2 * t + 1] = 1.0f - w0;
}

// deterministic compaction: lane e scans all tokens in order
__global__ void compact_kernel(const int* __restrict__ topk_idx,
                               const float* __restrict__ topk_w,
                               int* __restrict__ cnt, int* __restrict__ tlist,
                               float* __restrict__ wlist) {
  int e = threadIdx.x;
  if (e >= NEXP) return;
  int c = 0;
  for (int t = 0; t < NTOK; ++t) {
    int i0 = topk_idx[2 * t], i1 = topk_idx[2 * t + 1];
    if (i0 == e)      { tlist[e * NTOK + c] = t; wlist[e * NTOK + c] = topk_w[2 * t];     ++c; }
    else if (i1 == e) { tlist[e * NTOK + c] = t; wlist[e * NTOK + c] = topk_w[2 * t + 1]; ++c; }
  }
  cnt[e] = c;
}

// =======================================================================
// Fused expert MLP, all-bf16 operands:
//   Phase A: gate/up GEMM (WMMA) + SwiGLU -> swizzled h tiles in LDS
//   Phase B: h @ down^T (A-fragments = single 32B LDS reads) -> weighted scatter-add
extern __shared__ unsigned char smem_raw[];

__global__ __launch_bounds__(256) void moe_expert_kernel(
    const unsigned short* __restrict__ x16,    // [NTOK][HIDDEN] bf16
    const unsigned short* __restrict__ gu16,   // swizzled [E*2*INTER/16][HIDDEN/32] tiles
    const unsigned short* __restrict__ dn16,   // swizzled [E*HIDDEN/16][INTER/32] tiles
    const int* __restrict__ cnt,
    const int* __restrict__ tlist,
    const float* __restrict__ wlist,
    float* __restrict__ out) {
  const int tiles_per_e = NTOK / MTILE;
  const int e  = blockIdx.x / tiles_per_e;
  const int bt = blockIdx.x % tiles_per_e;
  int nrows = cnt[e] - bt * MTILE;
  if (nrows <= 0) return;
  if (nrows > MTILE) nrows = MTILE;

  unsigned short* xs = (unsigned short*)smem_raw;                 // [MTILE][XSTR]
  unsigned short* hs = xs + MTILE * XSTR;                         // 128 swizzled tiles
  int*   toks = (int*)(hs + 2 * (INTER / 32) * HTSTRIDE);
  float* wts  = (float*)(toks + MTILE);

  const int tid  = threadIdx.x;
  const int wave = tid >> 5;
  const int lane = tid & 31;

  if (tid < MTILE) {
    int idx = bt * MTILE + tid;
    if (tid < nrows) {
      toks[tid] = tlist[e * NTOK + idx];
      wts[tid]  = wlist[e * NTOK + idx];
    } else { toks[tid] = 0; wts[tid] = 0.0f; }   // padded rows read token 0, masked later
  }
  __syncthreads();

  // ---- gather 32 token rows into LDS via async global->LDS B128 (no VGPR round trip)
  {
    const unsigned xs_base = lds_off(xs);
    for (int i = tid; i < MTILE * (HIDDEN / 8); i += 256) {   // 16B chunks
      int m = i / (HIDDEN / 8);
      int c = (i % (HIDDEN / 8)) * 8;
      unsigned ldsa = xs_base + (unsigned)(m * XSTR + c) * 2u;
      const unsigned short* g = x16 + (size_t)toks[m] * HIDDEN + c;
      asm volatile("global_load_async_to_lds_b128 %0, %1, off"
                   :: "v"(ldsa), "v"(g) : "memory");
    }
    asm volatile("s_wait_asynccnt 0x0" ::: "memory");
  }
  __syncthreads();

  // ---- Phase A: gate/up GEMM + SwiGLU -> swizzled hs ----
  {
    const size_t erow = (size_t)e * (2 * INTER);
    for (int job = wave; job < 2 * (INTER / 16); job += NWAVES) {
      const int m0 = (job & 1) * 16;
      const int n0 = (job >> 1) * 16;
      const unsigned short* gbase = gu16 + ((erow + n0) >> 4) * (size_t)(HIDDEN / 32) * 512;
      const unsigned short* ubase = gu16 + ((erow + INTER + n0) >> 4) * (size_t)(HIDDEN / 32) * 512;
      v8f accg = {0.f, 0.f, 0.f, 0.f, 0.f, 0.f, 0.f, 0.f};
      v8f accu = {0.f, 0.f, 0.f, 0.f, 0.f, 0.f, 0.f, 0.f};
#pragma unroll 4
      for (int kt = 0; kt < HIDDEN / 32; ++kt) {
        v16bf a  = load_a_rowmajor(xs, XSTR, m0, kt * 32, lane);
        v16bf bg = frag_from(gbase + (size_t)kt * 512, lane);
        v16bf bu = frag_from(ubase + (size_t)kt * 512, lane);
        accg = wmma_bf16(a, bg, accg);
        accu = wmma_bf16(a, bu, accu);
      }
      // C layout: lane%16 = n; VGPR r -> row m0 + 8*(lane/16) + r. Scatter into
      // A-fragment-swizzled hs so Phase B reads one 32B vector per lane.
      const int n   = n0 + (lane & 15);
      const int kt  = n >> 5, kl = n & 31;
      const int lgc = (kl >> 3) & 1;                       // lane-group of K
      const int j   = (kl & 7) + ((kl >> 4) << 3);         // elem index in lane
      const int mlo = ((lane >> 4) << 3);                  // 0 or 8 within subtile
      unsigned short* tb = hs + (size_t)((m0 >> 4) * (INTER / 32) + kt) * HTSTRIDE;
#pragma unroll
      for (int r = 0; r < 8; ++r) {
        float g = accg[r];
        float u = fminf(accu[r], LIMIT);
        float s = g * __builtin_amdgcn_rcpf(1.0f + __expf(-g));   // silu
        tb[(mlo + r + 16 * lgc) * 16 + j] = f2bf(s * u);
      }
    }
  }
  __syncthreads();

  // ---- Phase B: out += (h @ down^T) * w ----
  {
    const size_t erow = (size_t)e * HIDDEN;
    for (int job = wave; job < 2 * (HIDDEN / 16); job += NWAVES) {
      const int m0 = (job & 1) * 16;
      const int n0 = (job >> 1) * 16;
      const unsigned short* dbase = dn16 + ((erow + n0) >> 4) * (size_t)(INTER / 32) * 512;
      const unsigned short* habase = hs + (size_t)(m0 >> 4) * (INTER / 32) * HTSTRIDE;
      v8f acc = {0.f, 0.f, 0.f, 0.f, 0.f, 0.f, 0.f, 0.f};
#pragma unroll 4
      for (int kt = 0; kt < INTER / 32; ++kt) {
        v16bf a = frag_from(habase + (size_t)kt * HTSTRIDE, lane);   // 32B LDS read
        v16bf b = frag_from(dbase + (size_t)kt * 512, lane);
        acc = wmma_bf16(a, b, acc);
      }
      const int n     = n0 + (lane & 15);
      const int mbase = m0 + ((lane >> 4) << 3);
#pragma unroll
      for (int r = 0; r < 8; ++r) {
        int m = mbase + r;
        if (m < nrows)
          atomicAdd(out + (size_t)toks[m] * HIDDEN + n, acc[r] * wts[m]);
      }
    }
  }
}

// =======================================================================
extern "C" void kernel_launch(void* const* d_in, const int* in_sizes, int n_in,
                              void* d_out, int out_size, void* d_ws, size_t ws_size,
                              hipStream_t stream) {
  const float* x  = (const float*)d_in[0];
  const float* rw = (const float*)d_in[1];
  const float* gu = (const float*)d_in[2];
  const float* dn = (const float*)d_in[3];
  float* out = (float*)d_out;

  // workspace layout (~86 MB): [0,1MB) routing; then bf16 x / swizzled bf16 weights
  char*  ws       = (char*)d_ws;
  int*   topk_idx = (int*)ws;
  float* topk_w   = (float*)(ws + 65536);
  int*   cnt      = (int*)(ws + 131072);
  int*   tlist    = (int*)(ws + 131072 + 128);
  float* wlist    = (float*)(ws + 131072 + 128 + 262144);
  unsigned short* x16  = (unsigned short*)(ws + (1u << 20));
  unsigned short* gu16 = (unsigned short*)(ws + (1u << 20) + (size_t)NTOK * HIDDEN * 2);
  unsigned short* dn16 = (unsigned short*)((char*)gu16 + (size_t)NEXP * 2 * INTER * HIDDEN * 2);

  zero_out_kernel<<<(NTOK * HIDDEN + 255) / 256, 256, 0, stream>>>(out, NTOK * HIDDEN);
  cvt_x_kernel<<<(NTOK * HIDDEN / 4 + 255) / 256, 256, 0, stream>>>(x, x16);
  {
    int jobs_gu = (NEXP * 2 * INTER / 16) * (HIDDEN / 32) * 32;
    swz_w_kernel<<<(jobs_gu + 255) / 256, 256, 0, stream>>>(gu, gu16, NEXP * 2 * INTER, HIDDEN);
    int jobs_dn = (NEXP * HIDDEN / 16) * (INTER / 32) * 32;
    swz_w_kernel<<<(jobs_dn + 255) / 256, 256, 0, stream>>>(dn, dn16, NEXP * HIDDEN, INTER);
  }
  router_kernel<<<NTOK / 256, 256, 0, stream>>>(x, rw, topk_idx, topk_w);
  compact_kernel<<<1, 32, 0, stream>>>(topk_idx, topk_w, cnt, tlist, wlist);

  const size_t lds_bytes = (size_t)MTILE * XSTR * 2                  // xs
                         + (size_t)2 * (INTER / 32) * HTSTRIDE * 2   // hs (swizzled)
                         + MTILE * 4 + MTILE * 4;                    // toks + wts
  moe_expert_kernel<<<NEXP * (NTOK / MTILE), 256, lds_bytes, stream>>>(
      x16, gu16, dn16, cnt, tlist, wlist, out);
}
